// RegionProposalNetwork_65274912964950
// MI455X (gfx1250) — matrix operand order repaired and checked by reading, not compiled, not run on hardware
//
#include <hip/hip_runtime.h>
#include <math.h>

// ---------------- types for WMMA / TDM (gfx1250, wave32) ----------------
typedef __attribute__((ext_vector_type(16))) __bf16   v16bf;
typedef __attribute__((ext_vector_type(8)))  float    v8f;
typedef __attribute__((ext_vector_type(4)))  unsigned u32x4;
typedef __attribute__((ext_vector_type(8)))  int      i32x8;
typedef __attribute__((ext_vector_type(4)))  int      i32x4;

union BfTile { v16bf v; u32x4 q[2]; };

#if defined(__has_builtin)
#if __has_builtin(__builtin_amdgcn_tensor_load_to_lds) && \
    __has_builtin(__builtin_amdgcn_s_wait_tensorcnt)
#define HAVE_TDM 1
#endif
#endif
#ifndef HAVE_TDM
#define HAVE_TDM 0
#endif

// ---------------- problem constants ----------------
#define BATCH   4
#define CIN     512
#define FH      50
#define FW      84
#define PH      52               // padded (halo) height
#define PW      86               // padded (halo) width
#define HW      (FH*FW)          // 4200
#define NPOS    (BATCH*HW)       // 16800
#define NANCH   (HW*9)           // 37800 per image
#define NTILES  (NPOS/16)        // 1050
#define NPOST   300
#define NPRE    3000
#define NEGINF  (-__builtin_huge_valf())

// d_out layout (floats): rpn_locs | rpn_scores | rois | roi_indices | anchors
#define O_LOCS   0
#define O_SCORES (BATCH*NANCH*4)                 // 604800
#define O_ROIS   (O_SCORES + BATCH*NANCH*2)      // 907200
#define O_RIDX   (O_ROIS + BATCH*NPOST*4)        // 912000
#define O_ANCH   (O_RIDX + BATCH*NPOST)          // 913200

// workspace layout (bytes, 256B aligned)
#define ALIGN256(x) (((x) + 255ull) & ~255ull)
#define XT_OFF    0ull                                        // x -> padded NHWC bf16
#define XT_BYTES  ((unsigned long long)BATCH*PH*PW*CIN*2)
#define WB_OFF    ALIGN256(XT_OFF + XT_BYTES)                 // w1 -> [oc][tap][ic] bf16
#define WB_BYTES  ((unsigned long long)CIN*9*CIN*2)
#define WSL_OFF   ALIGN256(WB_OFF + WB_BYTES)                 // head weights [64][512] bf16
#define WSL_BYTES (64ull*CIN*2)
#define CB_OFF    ALIGN256(WSL_OFF + WSL_BYTES)               // conv out [n][512] bf16
#define CB_BYTES  ((unsigned long long)NPOS*CIN*2)
#define FG_OFF    ALIGN256(CB_OFF + CB_BYTES)                 // fg scores fp32
#define FG_BYTES  ((unsigned long long)BATCH*NANCH*4)
#define ROI_OFF   ALIGN256(FG_OFF + FG_BYTES)                 // decoded rois fp32
#define ROI_BYTES ((unsigned long long)BATCH*NANCH*4*4)
#define SC_OFF    ALIGN256(ROI_OFF + ROI_BYTES)               // nms scores fp32
#define SC_BYTES  ((unsigned long long)BATCH*NANCH*4)

// ================= pre-pass repacks =================
// x: NCHW f32 -> xt: [b][py][px][c] bf16 with 1-pixel zero halo
__global__ void repack_x(const float* __restrict__ x, __bf16* __restrict__ xt) {
  int i = blockIdx.x * blockDim.x + threadIdx.x;
  if (i >= BATCH*PH*PW*CIN) return;
  int b = i / (PH*PW*CIN);
  int r = i % (PH*PW*CIN);
  int p = r / CIN;
  int c = r % CIN;
  int py = p / PW, px = p % PW;
  int iy = py - 1, ix = px - 1;
  float v = 0.f;
  if (iy >= 0 && iy < FH && ix >= 0 && ix < FW)
    v = x[((size_t)(b*CIN + c)*FH + iy)*FW + ix];
  xt[i] = (__bf16)v;
}

// w1: OIHW f32 -> wb: [oc][tap][ic] bf16
__global__ void repack_w1(const float* __restrict__ w1, __bf16* __restrict__ wb) {
  int i = blockIdx.x * blockDim.x + threadIdx.x;
  if (i >= CIN*CIN*9) return;
  int oc  = i / (CIN*9);
  int r   = i % (CIN*9);
  int ic  = r / 9;
  int tap = r % 9;
  wb[((size_t)oc*9 + tap)*CIN + ic] = (__bf16)w1[i];
}

// head weights -> wsl[64][512] bf16 (rows 0..17 score, 18..53 loc, 54..63 zero)
__global__ void repack_wh(const float* __restrict__ w_score,
                          const float* __restrict__ w_loc,
                          __bf16* __restrict__ wsl) {
  int i = blockIdx.x * blockDim.x + threadIdx.x;
  if (i >= 64*CIN) return;
  int m = i / CIN, ic = i % CIN;
  float v = 0.f;
  if (m < 18)       v = w_score[m*CIN + ic];
  else if (m < 54)  v = w_loc[(m-18)*CIN + ic];
  wsl[i] = (__bf16)v;
}

// ================= conv3x3 + bias + relu via WMMA bf16 =================
// grid (1050, 8), block 128 (4 waves); wave w -> mtile = by*4+w
// Halo-padded input => zero predication in the hot loop: pure loads + WMMA.
__global__ void conv3_wmma(const __bf16* __restrict__ xt,
                           const __bf16* __restrict__ wb,
                           const float* __restrict__ b1,
                           __bf16* __restrict__ convb) {
  const int lane  = threadIdx.x & 31;
  const int wave  = threadIdx.x >> 5;
  const int mtile = blockIdx.y * 4 + wave;   // 0..31
  const int ntile = blockIdx.x;              // 0..1049
  const int l15   = lane & 15;
  const bool hi   = lane >= 16;

  const int n   = ntile * 16 + l15;          // spatial column
  const int b   = n / HW;
  const int rem = n % HW;
  const int y   = rem / FW;
  const int x   = rem % FW;
  const int m   = mtile * 16 + l15;          // A row (out channel)

  v8f c = {};

  for (int tap = 0; tap < 9; ++tap) {
    const int py = y + tap / 3;              // padded coords (halo already applied)
    const int px = x + tap % 3;
    // A: lane holds M=l15, K in {kb..kb+7, kb+16..kb+23}, kb = hi?8:0
    const __bf16* aptr = wb + ((size_t)m * 9 + tap) * CIN + (hi ? 8 : 0);
    // B: lane holds N=l15, K in {kb..kb+7, kb+16..kb+23}, kb = hi?8:0
    const __bf16* bptr = xt + ((size_t)((b * PH + py) * PW + px)) * CIN + (hi ? 8 : 0);
    __builtin_prefetch(aptr, 0, 3);          // global_prefetch of weight stream
#pragma unroll 4
    for (int ic0 = 0; ic0 < CIN; ic0 += 32) {
      BfTile A, Bm;
      A.q[0]  = *reinterpret_cast<const u32x4*>(aptr + ic0);
      A.q[1]  = *reinterpret_cast<const u32x4*>(aptr + ic0 + 16);
      Bm.q[0] = *reinterpret_cast<const u32x4*>(bptr + ic0);
      Bm.q[1] = *reinterpret_cast<const u32x4*>(bptr + ic0 + 16);
      c = __builtin_amdgcn_wmma_f32_16x16x32_bf16(
              false, A.v, false, Bm.v, (short)0, c, false, false);
    }
  }

  // C layout: lane col = l15; rows = mbase..mbase+7
  const int mbase = mtile * 16 + (hi ? 8 : 0);
  __bf16* outp = convb + (size_t)n * CIN + mbase;
#pragma unroll
  for (int r = 0; r < 8; ++r) {
    float v = c[r] + b1[mbase + r];
    v = v > 0.f ? v : 0.f;                   // ReLU
    outp[r] = (__bf16)v;
  }
}

// ================= 1x1 heads via WMMA: scores(18) + locs(36) =================
// grid 1050, block 128 (4 waves); wave = mtile 0..3 covers m 0..63.
// The 16x512 bf16 activation tile is DMA'd to LDS once via the Tensor Data
// Mover and reused by all 4 waves / all 16 K-chunks (ds_load_b128 fragments).
__global__ void head_wmma(const __bf16* __restrict__ convb,
                          const __bf16* __restrict__ wsl,
                          const float* __restrict__ b_score,
                          const float* __restrict__ b_loc,
                          float* __restrict__ o_locs,
                          float* __restrict__ o_scores,
                          float* __restrict__ fg) {
  __shared__ __align__(128) unsigned char btile[16 * 1024];  // 16 rows x 1KB

  const int lane  = threadIdx.x & 31;
  const int mtile = threadIdx.x >> 5;
  const int ntile = blockIdx.x;
  const int l15   = lane & 15;
  const bool hi   = lane >= 16;

  const int n   = ntile * 16 + l15;
  const int b   = n / HW;
  const int pos = n % HW;

  const unsigned char* gsrc =
      (const unsigned char*)(convb + (size_t)(ntile * 16) * CIN);

#if HAVE_TDM
  if (threadIdx.x < 32) {  // wave 0 issues the TDM descriptor (EXEC ignored)
    unsigned long long ga = (unsigned long long)(uintptr_t)gsrc;
    unsigned lds_off = (unsigned)(uintptr_t)&btile[0];
    u32x4 g0;
    g0.x = 1u;                                     // count=1, user descriptor
    g0.y = lds_off;                                // lds_addr
    g0.z = (unsigned)ga;                           // global_addr[31:0]
    g0.w = ((unsigned)(ga >> 32) & 0x01FFFFFFu)    // global_addr[56:32]
           | 0x80000000u;                          // type=2 ("image")
    i32x8 g1;
    g1[0] = 0x00010000;        // wg_mask=0, data_size=1 (2B), no pad/iter
    g1[1] = (int)(512u << 16); // atomic_barrier_addr=0 | tensor_dim0.lo=512
    g1[2] = (int)(16u << 16);  // tensor_dim0.hi=0 | tensor_dim1.lo=16
    g1[3] = (int)(512u << 16); // tensor_dim1.hi=0 | tile_dim0=512
    g1[4] = 16;                // tile_dim1=16 | tile_dim2=0
    g1[5] = 512;               // tensor_dim0_stride.lo (elements)
    g1[6] = 0;                 // stride0.hi | stride1.lo
    g1[7] = 0;                 // stride1.hi
    i32x4 gz4 = {0, 0, 0, 0};  // groups 2/3 unused (2D tensor)
    i32x8 gz8 = {0, 0, 0, 0, 0, 0, 0, 0};
    // amdgpu-toolchain (clang-23) 6-arg form: g0, g1, g2, g3, extra group, cpol
    __builtin_amdgcn_tensor_load_to_lds(g0, g1, gz4, gz4, gz8, 0);
    __builtin_amdgcn_s_wait_tensorcnt(0);
  }
  __syncthreads();
#else
  for (int off = threadIdx.x * 16; off < 16 * 1024; off += blockDim.x * 16)
    *reinterpret_cast<u32x4*>(btile + off) =
        *reinterpret_cast<const u32x4*>(gsrc + off);
  __syncthreads();
#endif

  const __bf16* aptr = wsl + ((size_t)(mtile * 16 + l15)) * CIN + (hi ? 8 : 0);
  const unsigned char* bl = btile + l15 * 1024 + (hi ? 16 : 0);

  v8f c = {};
#pragma unroll 4
  for (int ic0 = 0; ic0 < CIN; ic0 += 32) {
    BfTile A, Bm;
    A.q[0]  = *reinterpret_cast<const u32x4*>(aptr + ic0);
    A.q[1]  = *reinterpret_cast<const u32x4*>(aptr + ic0 + 16);
    Bm.q[0] = *reinterpret_cast<const u32x4*>(bl + ic0 * 2);       // ds_load_b128
    Bm.q[1] = *reinterpret_cast<const u32x4*>(bl + ic0 * 2 + 32);
    c = __builtin_amdgcn_wmma_f32_16x16x32_bf16(
            false, A.v, false, Bm.v, (short)0, c, false, false);
  }

  const int mbase = mtile * 16 + (hi ? 8 : 0);   // always even
#pragma unroll
  for (int r = 0; r < 8; ++r) {
    const int mg = mbase + r;
    if (mg < 18) {
      const int a = mg >> 1, s = mg & 1;
      float v = c[r] + b_score[mg];
      o_scores[(size_t)b * (NANCH*2) + (pos * 9 + a) * 2 + s] = v;
      if (s == 0) {  // pair (mg, mg+1) lives in c[r], c[r+1]
        float v1 = c[r + 1] + b_score[mg + 1];
        fg[(size_t)b * NANCH + pos * 9 + a] = 1.f / (1.f + expf(v - v1));
      }
    } else if (mg < 54) {
      const int lm = mg - 18;
      const int a = lm >> 2, comp = lm & 3;
      float v = c[r] + b_loc[lm];
      o_locs[(size_t)b * (NANCH*4) + (pos * 9 + a) * 4 + comp] = v;
    }
  }
}

// ================= anchors + loc2bbox + clip + min-size =================
__global__ void proposals_prep(const float* __restrict__ o_locs,
                               const float* __restrict__ fg,
                               float* __restrict__ o_anch,
                               float* __restrict__ roi_ws,
                               float* __restrict__ sc_ws,
                               const int* __restrict__ img_h_p,
                               const int* __restrict__ img_w_p) {
  int i = blockIdx.x * blockDim.x + threadIdx.x;
  if (i >= NANCH) return;
  const int pos = i / 9, a = i % 9;
  const int y = pos / FW, x = pos % FW;
  const int ir = a / 3, is = a % 3;
  const float rr = (ir == 0) ? 0.5f : (ir == 1 ? 1.0f : 2.0f);
  const float ss = (is == 0) ? 8.f : (is == 1 ? 16.f : 32.f);
  const float ah = 16.f * ss * sqrtf(rr);
  const float aw = 16.f * ss * sqrtf(1.f / rr);
  const float shx = x * 16.f, shy = y * 16.f;
  const float a0 = shx - 0.5f * aw, a1 = shy - 0.5f * ah;
  const float a2 = shx + 0.5f * aw, a3 = shy + 0.5f * ah;
  o_anch[i * 4 + 0] = a0; o_anch[i * 4 + 1] = a1;
  o_anch[i * 4 + 2] = a2; o_anch[i * 4 + 3] = a3;

  const float AW = a2 - a0, AH = a3 - a1;
  const float acx = a0 + 0.5f * AW, acy = a1 + 0.5f * AH;
  const float img_w = (float)*img_w_p, img_h = (float)*img_h_p;

  for (int b = 0; b < BATCH; ++b) {
    const float* loc = o_locs + (size_t)b * (NANCH*4) + i * 4;
    float cx = loc[0] * AW + acx;
    float cy = loc[1] * AH + acy;
    float w  = expf(loc[2]) * AW;
    float h  = expf(loc[3]) * AH;
    float x1 = fminf(fmaxf(cx - 0.5f * w, 0.f), img_w);
    float y1 = fminf(fmaxf(cy - 0.5f * h, 0.f), img_h);
    float x2 = fminf(fmaxf(cx + 0.5f * w, 0.f), img_w);
    float y2 = fminf(fmaxf(cy + 0.5f * h, 0.f), img_h);
    bool ok = ((x2 - x1) >= 16.f) & ((y2 - y1) >= 16.f);
    float* R = roi_ws + ((size_t)b * NANCH + i) * 4;
    R[0] = x1; R[1] = y1; R[2] = x2; R[3] = y2;
    sc_ws[(size_t)b * NANCH + i] = ok ? fg[(size_t)b * NANCH + i] : NEGINF;
  }
}

// ================= top-3000 threshold mask (binary search on float key order) ==
__device__ __forceinline__ unsigned score_key(float f) {
  unsigned u = __float_as_uint(f);
  return (u & 0x80000000u) ? ~u : (u | 0x80000000u);
}

__global__ void topk_mask(float* __restrict__ sc_ws) {
  const int b = blockIdx.x;
  float* sc = sc_ws + (size_t)b * NANCH;
  __shared__ unsigned cnt_s;
  unsigned lo = 0u, hi = 0xFFFFFFFFu;
  while (lo < hi) {
    unsigned long long span = (unsigned long long)hi - lo + 1ull;
    unsigned mid = lo + (unsigned)(span >> 1);
    if (threadIdx.x == 0) cnt_s = 0;
    __syncthreads();
    unsigned local = 0;
    for (int i = threadIdx.x; i < NANCH; i += blockDim.x)
      if (score_key(sc[i]) >= mid) local++;
    atomicAdd(&cnt_s, local);
    __syncthreads();
    unsigned c = cnt_s;
    __syncthreads();
    if (c >= NPRE) lo = mid; else hi = mid - 1;
  }
  for (int i = threadIdx.x; i < NANCH; i += blockDim.x)
    if (score_key(sc[i]) < lo) sc[i] = NEGINF;
}

// ================= greedy sequential NMS (one workgroup per image) ===========
__global__ void nms_kernel(float* __restrict__ sc_ws,
                           const float* __restrict__ roi_ws,
                           float* __restrict__ o_rois,
                           float* __restrict__ o_ridx) {
  const int b = blockIdx.x;
  float* sc = sc_ws + (size_t)b * NANCH;
  const float* bx = roi_ws + (size_t)b * NANCH * 4;
  __shared__ float sval[256];
  __shared__ int   sidx[256];
  __shared__ float box_s[5];

  for (int it = 0; it < NPOST; ++it) {
    // parallel argmax, tie-break on lowest index (matches jnp.argmax)
    float bestv = NEGINF; int besti = 0x7FFFFFFF;
    for (int i = threadIdx.x; i < NANCH; i += 256) {
      float v = sc[i];
      if (v > bestv || (v == bestv && i < besti)) { bestv = v; besti = i; }
    }
    sval[threadIdx.x] = bestv; sidx[threadIdx.x] = besti;
    __syncthreads();
    for (int off = 128; off > 0; off >>= 1) {
      if ((int)threadIdx.x < off) {
        float ov = sval[threadIdx.x + off]; int oi = sidx[threadIdx.x + off];
        if (ov > sval[threadIdx.x] ||
            (ov == sval[threadIdx.x] && oi < sidx[threadIdx.x])) {
          sval[threadIdx.x] = ov; sidx[threadIdx.x] = oi;
        }
      }
      __syncthreads();
    }
    if (threadIdx.x == 0) {
      const int idx = sidx[0];
      const bool valid = sval[0] > NEGINF;
      const float x1 = bx[idx*4+0], y1 = bx[idx*4+1];
      const float x2 = bx[idx*4+2], y2 = bx[idx*4+3];
      box_s[0] = x1; box_s[1] = y1; box_s[2] = x2; box_s[3] = y2;
      box_s[4] = (x2 - x1) * (y2 - y1);
      float* ro = o_rois + ((size_t)b * NPOST + it) * 4;
      ro[0] = valid ? x1 : 0.f; ro[1] = valid ? y1 : 0.f;
      ro[2] = valid ? x2 : 0.f; ro[3] = valid ? y2 : 0.f;
      o_ridx[b * NPOST + it] = (float)b;
      sc[idx] = NEGINF;
    }
    __syncthreads();
    const float X1 = box_s[0], Y1 = box_s[1], X2 = box_s[2], Y2 = box_s[3];
    const float AR = box_s[4];
    for (int i = threadIdx.x; i < NANCH; i += 256) {
      float bx1 = bx[i*4+0], by1 = bx[i*4+1], bx2 = bx[i*4+2], by2 = bx[i*4+3];
      float xx1 = fmaxf(X1, bx1), yy1 = fmaxf(Y1, by1);
      float xx2 = fminf(X2, bx2), yy2 = fminf(Y2, by2);
      float inter = fmaxf(xx2 - xx1, 0.f) * fmaxf(yy2 - yy1, 0.f);
      float iou = inter / (AR + (bx2 - bx1) * (by2 - by1) - inter);
      if (iou > 0.7f) sc[i] = NEGINF;
    }
    __syncthreads();
  }
}

// ================= launcher =================
extern "C" void kernel_launch(void* const* d_in, const int* in_sizes, int n_in,
                              void* d_out, int out_size, void* d_ws, size_t ws_size,
                              hipStream_t stream) {
  const float* x       = (const float*)d_in[0];
  const float* w1      = (const float*)d_in[1];
  const float* b1      = (const float*)d_in[2];
  const float* w_score = (const float*)d_in[3];
  const float* b_score = (const float*)d_in[4];
  const float* w_loc   = (const float*)d_in[5];
  const float* b_loc   = (const float*)d_in[6];
  const int*   img_h   = (const int*)d_in[7];
  const int*   img_w   = (const int*)d_in[8];

  char* ws = (char*)d_ws;
  __bf16* xt    = (__bf16*)(ws + XT_OFF);
  __bf16* wb    = (__bf16*)(ws + WB_OFF);
  __bf16* wsl   = (__bf16*)(ws + WSL_OFF);
  __bf16* convb = (__bf16*)(ws + CB_OFF);
  float*  fg    = (float*)(ws + FG_OFF);
  float*  roiws = (float*)(ws + ROI_OFF);
  float*  scws  = (float*)(ws + SC_OFF);

  float* out      = (float*)d_out;
  float* o_locs   = out + O_LOCS;
  float* o_scores = out + O_SCORES;
  float* o_rois   = out + O_ROIS;
  float* o_ridx   = out + O_RIDX;
  float* o_anch   = out + O_ANCH;

  repack_x <<<(BATCH*PH*PW*CIN + 255) / 256, 256, 0, stream>>>(x, xt);
  repack_w1<<<(CIN*CIN*9      + 255) / 256, 256, 0, stream>>>(w1, wb);
  repack_wh<<<(64*CIN         + 255) / 256, 256, 0, stream>>>(w_score, w_loc, wsl);

  conv3_wmma<<<dim3(NTILES, 8), 128, 0, stream>>>(xt, wb, b1, convb);
  head_wmma <<<dim3(NTILES),    128, 0, stream>>>(convb, wsl, b_score, b_loc,
                                                  o_locs, o_scores, fg);
  proposals_prep<<<(NANCH + 255) / 256, 256, 0, stream>>>(
      o_locs, fg, o_anch, roiws, scws, img_h, img_w);
  topk_mask <<<BATCH, 256, 0, stream>>>(scws);
  nms_kernel<<<BATCH, 256, 0, stream>>>(scws, roiws, o_rois, o_ridx);
}
